// TaggingItems_27410481283884
// MI455X (gfx1250) — matrix operand order repaired and checked by reading, not compiled, not run on hardware
//
#include <hip/hip_runtime.h>
#include <stdint.h>

// ---------------------------------------------------------------------------
// Problem constants (from the reference)
// ---------------------------------------------------------------------------
#define H        300     // hidden size
#define KP       320     // K padded to multiple of 32 (10 k-tiles of 32)
#define KT       10      // number of 32-wide K tiles
#define NG       1200    // gate width = 4*H
#define NGT      75      // gate column tiles of 16
#define NOT      19      // output column tiles of 16 (300 -> 304)
#define N_ITEM   50000
#define N_TAG    5000
#define MAX_DEG  64
#define NQ       65536   // number of queried items
#define NEG_SLOPE 0.01f

typedef __attribute__((ext_vector_type(8)))  __bf16 v8bf;
typedef __attribute__((ext_vector_type(16))) __bf16 v16bf;
typedef __attribute__((ext_vector_type(8)))  float  v8f;

// ---------------------------------------------------------------------------
// Helpers
// ---------------------------------------------------------------------------
static __device__ __forceinline__ unsigned short f2bf(float f) {
  union { float f; unsigned u; } x; x.f = f;
  unsigned r = x.u + 0x7FFFu + ((x.u >> 16) & 1u);   // round-to-nearest-even
  return (unsigned short)(r >> 16);
}
static __device__ __forceinline__ float bf2f(unsigned short h) {
  union { unsigned u; float f; } x; x.u = ((unsigned)h) << 16; return x.f;
}
static __device__ __forceinline__ float sigmoidf(float x) {
  return 1.0f / (1.0f + __expf(-x));
}

// A-fragment (16x32 bf16, row-major source with row stride KP elements).
// ISA 16-bit A layout: lane L -> M = L&15, khalf = L>>4;
//   v16 elements 0..7  = K = kt*32 + khalf*8 + e
//   v16 elements 8..15 = K = kt*32 + 16 + khalf*8 + (e-8)
// i.e. two contiguous 8-element (16B) chunks per lane.
static __device__ __forceinline__ v16bf load_a_frag(const unsigned short* rowbase,
                                                    int kt, int lane) {
  int khalf = lane >> 4;
  int o0 = kt * 32 + khalf * 8;
  v8bf c0 = *reinterpret_cast<const v8bf*>(rowbase + o0);
  v8bf c1 = *reinterpret_cast<const v8bf*>(rowbase + o0 + 16);
  return __builtin_shufflevector(c0, c1, 0,1,2,3,4,5,6,7,8,9,10,11,12,13,14,15);
}

// B-fragment: packed linear, one 32B contiguous chunk per lane.
static __device__ __forceinline__ v16bf load_b_frag(const unsigned short* Wp,
                                                    int kt, int nt, int ntiles, int lane) {
  const unsigned short* p = Wp + ((size_t)(kt * ntiles + nt) * 32 + lane) * 16;
  v8bf c0 = *reinterpret_cast<const v8bf*>(p);
  v8bf c1 = *reinterpret_cast<const v8bf*>(p + 8);
  return __builtin_shufflevector(c0, c1, 0,1,2,3,4,5,6,7,8,9,10,11,12,13,14,15);
}

// ---------------------------------------------------------------------------
// Weight packing: row-major f32 W[K][N] -> bf16 fragment-linear
//   Wp[kt][nt][lane][16], lane: n = nt*16 + (lane&15), K = kt*32 + (lane>>4)*8 + e + (e>=8 ? 8 : 0)
// ---------------------------------------------------------------------------
__global__ void pack_b_kernel(const float* __restrict__ W, unsigned short* __restrict__ Wp,
                              int K, int N, int ktiles, int ntiles) {
  int idx = blockIdx.x * blockDim.x + threadIdx.x;  // one thread per (frag, lane)
  int total = ktiles * ntiles * 32;
  if (idx >= total) return;
  int lane = idx & 31;
  int frag = idx >> 5;
  int nt = frag % ntiles;
  int kt = frag / ntiles;
  int n = nt * 16 + (lane & 15);
  int khalf = lane >> 4;
  unsigned short* dst = Wp + (size_t)idx * 16;
#pragma unroll
  for (int e = 0; e < 16; ++e) {
    int k = kt * 32 + khalf * 8 + e + (e >= 8 ? 8 : 0);
    float v = (k < K && n < N) ? W[(size_t)k * N + n] : 0.0f;
    dst[e] = f2bf(v);
  }
}

// f32 [rows][H] -> bf16 [rows][KP] (zero-padded K tail)
__global__ void cvt_pad_kernel(const float* __restrict__ src,
                               unsigned short* __restrict__ dst, int rows) {
  size_t idx = (size_t)blockIdx.x * blockDim.x + threadIdx.x;
  size_t total = (size_t)rows * KP;
  if (idx >= total) return;
  int col = (int)(idx % KP);
  size_t row = idx / KP;
  dst[idx] = (col < H) ? f2bf(src[row * H + col]) : (unsigned short)0;
}

// ---------------------------------------------------------------------------
// Fused LSTM neighbor aggregation.
// One workgroup owns 16 destination nodes and runs all 64 recurrence steps
// with h (bf16), c (f32) and the 16x1200 f32 gate block resident in LDS
// (~114 KB -- fits only because CDNA5 has 320 KB LDS per WGP).
// Per step: gather x_t rows -> LDS, then gates = x@Wih + h@Whh + b via
// v_wmma_f32_16x16x32_bf16, then the pointwise cell update.
//
// IMPORTANT: the t and kt loops must NOT be unrolled. If kt is unrolled,
// every B-fragment load becomes t-invariant, LICM hoists ~200 fragments
// (1600 VGPRs worth) out of the 64-step loop and the allocator spills the
// world to scratch (observed in round-1 asm). Keeping kt a real loop keeps
// the weight stream flowing from L2 each step with a ~130-VGPR live set.
// ---------------------------------------------------------------------------
__global__ __launch_bounds__(256)
void lstm_agg_kernel(const unsigned short* __restrict__ hSrc,   // [*, KP] bf16
                     const int*            __restrict__ nbr,    // [Nn, 64]
                     const unsigned char*  __restrict__ mask,   // [Nn, 64]
                     const unsigned short* __restrict__ WihP,   // packed [KT][NGT][32][16]
                     const unsigned short* __restrict__ WhhP,
                     const float*          __restrict__ bias,   // [1200]
                     unsigned short*       __restrict__ hnOut,  // [Nn, KP] bf16
                     int Nn) {
  extern __shared__ char smem[];
  unsigned short* xA = (unsigned short*)smem;            // 16*KP bf16   (10240 B)
  unsigned short* hA = xA + 16 * KP;                     // 16*KP bf16   (10240 B)
  float* cS = (float*)(hA + 16 * KP);                    // 16*H  f32    (19200 B)
  float* gS = cS + 16 * H;                               // 16*NG f32    (76800 B)
  int* srcS = (int*)(gS + 16 * NG);                      // 16 ints
  int* hasS = srcS + 16;                                 // 16 ints

  const int tid  = threadIdx.x;
  const int lane = tid & 31;
  const int wave = tid >> 5;
  const int row0 = blockIdx.x * 16;

  // init: h = 0 (incl. K padding -> stays 0 forever), c = 0, has-neighbor flags
  for (int i = tid; i < 16 * KP; i += 256) hA[i] = 0;
  for (int i = tid; i < 16 * H;  i += 256) cS[i] = 0.0f;
  if (tid < 16) {
    int node = min(row0 + tid, Nn - 1);
    int h = 0;
    for (int t = 0; t < MAX_DEG; ++t) h |= mask[(size_t)node * MAX_DEG + t];
    hasS[tid] = h;
  }
  __syncthreads();

  // per-wave gate column tiles: nt = wave + 8*j  (75 tiles over 8 waves)
  const int ntc = (wave < (NGT & 7)) ? (NGT / 8 + 1) : (NGT / 8);
  const int colb = lane & 15;
  float bb[10];
#pragma unroll
  for (int j = 0; j < 10; ++j)
    if (j < ntc) bb[j] = bias[(wave + 8 * j) * 16 + colb];

  const int arow = lane & 15;

#pragma clang loop unroll(disable)
  for (int t = 0; t < MAX_DEG; ++t) {
    // ---- stage x_t (gathered neighbor rows) into LDS -----------------
    if (tid < 16) {
      int node = min(row0 + tid, Nn - 1);
      srcS[tid] = nbr[(size_t)node * MAX_DEG + t];
    }
    __syncthreads();
    {
      // 16 rows * 40 16B-chunks = 640 chunks; thread handles i, i+256, i+512.
      // Issue all loads first, then all LDS stores, so the clause overlaps
      // the (L2-resident) gather latency.
      const int i0 = tid, i1 = tid + 256, i2 = tid + 512;
      const int r0 = i0 / (KP / 8), q0 = i0 - r0 * (KP / 8);
      const int r1 = i1 / (KP / 8), q1 = i1 - r1 * (KP / 8);
      const int r2 = i2 / (KP / 8), q2 = i2 - r2 * (KP / 8);
      uint4 a0 = reinterpret_cast<const uint4*>(hSrc + (size_t)srcS[r0] * KP)[q0];
      uint4 a1 = reinterpret_cast<const uint4*>(hSrc + (size_t)srcS[r1] * KP)[q1];
      uint4 a2 = (i2 < 640)
          ? reinterpret_cast<const uint4*>(hSrc + (size_t)srcS[r2] * KP)[q2]
          : (uint4){0, 0, 0, 0};
      reinterpret_cast<uint4*>(xA + (size_t)r0 * KP)[q0] = a0;
      reinterpret_cast<uint4*>(xA + (size_t)r1 * KP)[q1] = a1;
      if (i2 < 640) reinterpret_cast<uint4*>(xA + (size_t)r2 * KP)[q2] = a2;
    }
    __syncthreads();

    // ---- gates = x@Wih + h@Whh + b  (WMMA bf16 -> f32) ----------------
    v8f acc[10];
#pragma unroll
    for (int j = 0; j < 10; ++j)
      if (j < ntc) { float b = bb[j]; acc[j] = (v8f){b, b, b, b, b, b, b, b}; }

#pragma clang loop unroll(disable)
    for (int kt = 0; kt < KT; ++kt) {
      v16bf ax = load_a_frag(xA + (size_t)arow * KP, kt, lane);
      v16bf ah = load_a_frag(hA + (size_t)arow * KP, kt, lane);
      if (kt + 1 < KT) {   // pull next K-tile toward L2/L0 (global_prefetch_b8)
        __builtin_prefetch(WihP + ((size_t)((kt + 1) * NGT + wave) * 32 + lane) * 16, 0, 1);
        __builtin_prefetch(WhhP + ((size_t)((kt + 1) * NGT + wave) * 32 + lane) * 16, 0, 1);
      }
#pragma unroll
      for (int j = 0; j < 10; ++j) {
        if (j < ntc) {
          int nt = wave + 8 * j;
          v16bf bih = load_b_frag(WihP, kt, nt, NGT, lane);
          v16bf bhh = load_b_frag(WhhP, kt, nt, NGT, lane);
          acc[j] = __builtin_amdgcn_wmma_f32_16x16x32_bf16(false, ax, false, bih,
                                                           (short)0, acc[j], false, false);
          acc[j] = __builtin_amdgcn_wmma_f32_16x16x32_bf16(false, ah, false, bhh,
                                                           (short)0, acc[j], false, false);
        }
      }
    }

    // ---- spill gates to LDS (C layout: VGPR v -> M=v+(lane>=16?8:0)) --
    const int rofs = (lane >> 4) * 8;
#pragma unroll
    for (int j = 0; j < 10; ++j) {
      if (j < ntc) {
        int col = (wave + 8 * j) * 16 + colb;
#pragma unroll
        for (int v = 0; v < 8; ++v) gS[(size_t)(v + rofs) * NG + col] = acc[j][v];
      }
    }
    __syncthreads();

    // ---- pointwise LSTM cell update (masked) --------------------------
    for (int e = tid; e < 16 * H; e += 256) {
      int r = e / H;
      int c = e - r * H;
      int node = min(row0 + r, Nn - 1);
      float gi = sigmoidf(gS[(size_t)r * NG + c]);
      float gf = sigmoidf(gS[(size_t)r * NG + H + c]);
      float gg = tanhf   (gS[(size_t)r * NG + 2 * H + c]);
      float go = sigmoidf(gS[(size_t)r * NG + 3 * H + c]);
      float cOld = cS[e];
      float hOld = bf2f(hA[(size_t)r * KP + c]);
      float cNew = gf * cOld + gi * gg;
      float hNew = go * tanhf(cNew);
      bool m = mask[(size_t)node * MAX_DEG + t] != 0;
      cS[e] = m ? cNew : cOld;
      hA[(size_t)r * KP + c] = f2bf(m ? hNew : hOld);
    }
    __syncthreads();
  }

  // ---- write hn (zeroed when node has no neighbors), padded bf16 ------
  for (int i = tid; i < 16 * KP; i += 256) {
    int r = i / KP;
    int c = i - r * KP;
    int node = row0 + r;
    if (node < Nn) {
      unsigned short v = (hasS[r] && c < H) ? hA[i] : (unsigned short)0;
      hnOut[(size_t)node * KP + c] = v;
    }
  }
}

// ---------------------------------------------------------------------------
// out = hDst @ Wself + bself + hn @ Wneigh + bneigh   (f32 output [Nn,H])
// ---------------------------------------------------------------------------
__global__ __launch_bounds__(256)
void out_gemm_kernel(const unsigned short* __restrict__ hDst,   // [Nn,KP]
                     const unsigned short* __restrict__ hn,     // [Nn,KP]
                     const unsigned short* __restrict__ WselfP, // [KT][NOT][32][16]
                     const unsigned short* __restrict__ WneighP,
                     const float* __restrict__ bself,
                     const float* __restrict__ bneigh,
                     float* __restrict__ out, int Nn) {
  const int tid = threadIdx.x, lane = tid & 31, wave = tid >> 5;
  const int row0 = blockIdx.x * 16;
  const int arow = min(row0 + (lane & 15), Nn - 1);
  const unsigned short* rowD = hDst + (size_t)arow * KP;
  const unsigned short* rowN = hn   + (size_t)arow * KP;
  const int ntc = (wave < (NOT & 7)) ? (NOT / 8 + 1) : (NOT / 8);  // 19 tiles / 8 waves
  const int colb = lane & 15;

  v8f acc[3];
#pragma unroll
  for (int j = 0; j < 3; ++j)
    if (j < ntc) {
      int col = (wave + 8 * j) * 16 + colb;
      float b = (col < H) ? (bself[col] + bneigh[col]) : 0.0f;
      acc[j] = (v8f){b, b, b, b, b, b, b, b};
    }

#pragma clang loop unroll(disable)
  for (int kt = 0; kt < KT; ++kt) {
    v16bf ad = load_a_frag(rowD, kt, lane);
    v16bf an = load_a_frag(rowN, kt, lane);
#pragma unroll
    for (int j = 0; j < 3; ++j) {
      if (j < ntc) {
        int nt = wave + 8 * j;
        v16bf bs = load_b_frag(WselfP,  kt, nt, NOT, lane);
        v16bf bn = load_b_frag(WneighP, kt, nt, NOT, lane);
        acc[j] = __builtin_amdgcn_wmma_f32_16x16x32_bf16(false, ad, false, bs,
                                                         (short)0, acc[j], false, false);
        acc[j] = __builtin_amdgcn_wmma_f32_16x16x32_bf16(false, an, false, bn,
                                                         (short)0, acc[j], false, false);
      }
    }
  }

  const int rofs = (lane >> 4) * 8;
#pragma unroll
  for (int j = 0; j < 3; ++j) {
    if (j < ntc) {
      int col = (wave + 8 * j) * 16 + colb;
      if (col < H) {
#pragma unroll
        for (int v = 0; v < 8; ++v) {
          int node = row0 + rofs + v;
          if (node < Nn) out[(size_t)node * H + col] = acc[j][v];
        }
      }
    }
  }
}

// ---------------------------------------------------------------------------
// final = leaky_relu(h_item @ Wfinal)[items]  -- only the 65536 queried rows.
// ---------------------------------------------------------------------------
__global__ __launch_bounds__(256)
void final_gemm_kernel(const unsigned short* __restrict__ hItem, // [N_ITEM,KP]
                       const int* __restrict__ items,            // [NQ]
                       const unsigned short* __restrict__ WfinP, // [KT][NOT][32][16]
                       float* __restrict__ out) {                // [NQ,H]
  const int tid = threadIdx.x, lane = tid & 31, wave = tid >> 5;
  const int row0 = blockIdx.x * 16;                 // NQ % 16 == 0
  const int r = row0 + (lane & 15);
  const unsigned short* rowA = hItem + (size_t)items[r] * KP;
  const int ntc = (wave < (NOT & 7)) ? (NOT / 8 + 1) : (NOT / 8);
  const int colb = lane & 15;

  v8f acc[3];
#pragma unroll
  for (int j = 0; j < 3; ++j)
    if (j < ntc) acc[j] = (v8f){0, 0, 0, 0, 0, 0, 0, 0};

#pragma clang loop unroll(disable)
  for (int kt = 0; kt < KT; ++kt) {
    v16bf a = load_a_frag(rowA, kt, lane);
#pragma unroll
    for (int j = 0; j < 3; ++j) {
      if (j < ntc) {
        int nt = wave + 8 * j;
        v16bf b = load_b_frag(WfinP, kt, nt, NOT, lane);
        acc[j] = __builtin_amdgcn_wmma_f32_16x16x32_bf16(false, a, false, b,
                                                         (short)0, acc[j], false, false);
      }
    }
  }

  const int rofs = (lane >> 4) * 8;
#pragma unroll
  for (int j = 0; j < 3; ++j) {
    if (j < ntc) {
      int col = (wave + 8 * j) * 16 + colb;
      if (col < H) {
#pragma unroll
        for (int v = 0; v < 8; ++v) {
          int rr = row0 + rofs + v;
          float x = acc[j][v];
          out[(size_t)rr * H + col] = (x > 0.0f) ? x : NEG_SLOPE * x;
        }
      }
    }
  }
}

// ---------------------------------------------------------------------------
// LayerNorm over H, writes next-layer bf16 state directly (padded).
// ---------------------------------------------------------------------------
__global__ __launch_bounds__(320)
void layernorm_kernel(const float* __restrict__ x, const float* __restrict__ g,
                      const float* __restrict__ b, unsigned short* __restrict__ hOut) {
  __shared__ float s1[320], s2[320];
  const int row = blockIdx.x;
  const int t = threadIdx.x;
  float v = (t < H) ? x[(size_t)row * H + t] : 0.0f;
  s1[t] = v; s2[t] = v * v;
  __syncthreads();
  for (int s = 256; s > 0; s >>= 1) {
    if (t < s && t + s < 320) { s1[t] += s1[t + s]; s2[t] += s2[t + s]; }
    __syncthreads();
  }
  float mu = s1[0] / (float)H;
  float var = s2[0] / (float)H - mu * mu;
  float rstd = rsqrtf(var + 1e-5f);
  unsigned short o = 0;
  if (t < H) o = f2bf((v - mu) * rstd * g[t] + b[t]);
  hOut[(size_t)row * KP + t] = o;
}

// ---------------------------------------------------------------------------
// Host launcher
// ---------------------------------------------------------------------------
extern "C" void kernel_launch(void* const* d_in, const int* in_sizes, int n_in,
                              void* d_out, int out_size, void* d_ws, size_t ws_size,
                              hipStream_t stream) {
  (void)in_sizes; (void)n_in; (void)out_size; (void)ws_size;

  const float* item_emb = (const float*)d_in[0];
  const float* tag_emb  = (const float*)d_in[1];
  const float* Wfinal   = (const float*)d_in[66];
  const int*   items    = (const int*)d_in[67];
  const int*   tag_nbr  = (const int*)d_in[68];
  const unsigned char* tag_mask  = (const unsigned char*)d_in[69];
  const int*   item_nbr = (const int*)d_in[70];
  const unsigned char* item_mask = (const unsigned char*)d_in[71];

  // Workspace carving
  char* ws = (char*)d_ws;
  size_t off = 0;
  auto alloc = [&](size_t bytes) -> char* {
    char* p = ws + off; off += (bytes + 255) & ~(size_t)255; return p;
  };

  const size_t GATE_PACK = (size_t)KT * NGT * 32 * 16 * sizeof(unsigned short); // 768000 B
  const size_t OUT_PACK  = (size_t)KT * NOT * 32 * 16 * sizeof(unsigned short); // 194560 B

  unsigned short *WihP[8], *WhhP[8], *WselfP[8], *WneighP[8];
  for (int d = 0; d < 8; ++d) {
    WihP[d]    = (unsigned short*)alloc(GATE_PACK);
    WhhP[d]    = (unsigned short*)alloc(GATE_PACK);
    WselfP[d]  = (unsigned short*)alloc(OUT_PACK);
    WneighP[d] = (unsigned short*)alloc(OUT_PACK);
  }
  unsigned short* WfinP = (unsigned short*)alloc(OUT_PACK);
  unsigned short* hI  = (unsigned short*)alloc((size_t)N_ITEM * KP * 2);
  unsigned short* hT  = (unsigned short*)alloc((size_t)N_TAG  * KP * 2);
  unsigned short* hnI = (unsigned short*)alloc((size_t)N_ITEM * KP * 2);
  unsigned short* hnT = (unsigned short*)alloc((size_t)N_TAG  * KP * 2);
  float* newI = (float*)alloc((size_t)N_ITEM * H * 4);
  float* newT = (float*)alloc((size_t)N_TAG  * H * 4);

  // ---- pack all weights once ------------------------------------------
  const int gateThreads = KT * NGT * 32;   // 24000
  const int outThreads  = KT * NOT * 32;   // 6080
  for (int l = 0; l < 4; ++l) {
    for (int et = 0; et < 2; ++et) {       // 0 = 'as' (tags), 1 = 'ras' (items)
      int base = 2 + l * 16 + et * 7;
      int d = l * 2 + et;
      pack_b_kernel<<<(gateThreads + 255) / 256, 256, 0, stream>>>(
          (const float*)d_in[base + 0], WihP[d], H, NG, KT, NGT);
      pack_b_kernel<<<(gateThreads + 255) / 256, 256, 0, stream>>>(
          (const float*)d_in[base + 1], WhhP[d], H, NG, KT, NGT);
      pack_b_kernel<<<(outThreads + 255) / 256, 256, 0, stream>>>(
          (const float*)d_in[base + 3], WselfP[d], H, H, KT, NOT);
      pack_b_kernel<<<(outThreads + 255) / 256, 256, 0, stream>>>(
          (const float*)d_in[base + 4], WneighP[d], H, H, KT, NOT);
    }
  }
  pack_b_kernel<<<(outThreads + 255) / 256, 256, 0, stream>>>(
      Wfinal, WfinP, H, H, KT, NOT);

  // ---- initial embeddings -> bf16 padded ------------------------------
  {
    size_t nI = (size_t)N_ITEM * KP, nT = (size_t)N_TAG * KP;
    cvt_pad_kernel<<<(unsigned)((nI + 255) / 256), 256, 0, stream>>>(item_emb, hI, N_ITEM);
    cvt_pad_kernel<<<(unsigned)((nT + 255) / 256), 256, 0, stream>>>(tag_emb, hT, N_TAG);
  }

  const size_t LSTM_LDS = (size_t)16 * KP * 2 * 2      // xA + hA (bf16)
                        + (size_t)16 * H * 4           // c (f32)
                        + (size_t)16 * NG * 4          // gates (f32)
                        + 32 * sizeof(int);            // srcS + hasS
  const int tagBlocks  = (N_TAG  + 15) / 16;   // 313
  const int itemBlocks = (N_ITEM + 15) / 16;   // 3125

  // ---- 4 layers --------------------------------------------------------
  for (int l = 0; l < 4; ++l) {
    int das = l * 2 + 0, dras = l * 2 + 1;
    int base_as  = 2 + l * 16;
    int base_ras = 2 + l * 16 + 7;

    // new_tag = SAGE(h_item -> tags)
    lstm_agg_kernel<<<tagBlocks, 256, LSTM_LDS, stream>>>(
        hI, tag_nbr, tag_mask, WihP[das], WhhP[das],
        (const float*)d_in[base_as + 2], hnT, N_TAG);
    out_gemm_kernel<<<tagBlocks, 256, 0, stream>>>(
        hT, hnT, WselfP[das], WneighP[das],
        (const float*)d_in[base_as + 5], (const float*)d_in[base_as + 6], newT, N_TAG);

    // new_item = SAGE(h_tag -> items)
    lstm_agg_kernel<<<itemBlocks, 256, LSTM_LDS, stream>>>(
        hT, item_nbr, item_mask, WihP[dras], WhhP[dras],
        (const float*)d_in[base_ras + 2], hnI, N_ITEM);
    out_gemm_kernel<<<itemBlocks, 256, 0, stream>>>(
        hI, hnI, WselfP[dras], WneighP[dras],
        (const float*)d_in[base_ras + 5], (const float*)d_in[base_ras + 6], newI, N_ITEM);

    // LayerNorm(item) -> next-layer bf16 state; tag passthrough -> bf16
    layernorm_kernel<<<N_ITEM, 320, 0, stream>>>(
        newI, (const float*)d_in[2 + l * 16 + 14], (const float*)d_in[2 + l * 16 + 15], hI);
    {
      size_t nT = (size_t)N_TAG * KP;
      cvt_pad_kernel<<<(unsigned)((nT + 255) / 256), 256, 0, stream>>>(newT, hT, N_TAG);
    }
  }

  // ---- final projection + leaky relu on the queried rows --------------
  final_gemm_kernel<<<NQ / 16, 256, 0, stream>>>(hI, items, WfinP, (float*)d_out);
}
